// AttentionSequencePoolingLayer_89730456748771
// MI455X (gfx1250) — matrix operand order repaired and checked by reading, not compiled, not run on hardware
//
#include <hip/hip_runtime.h>
#include <hip/hip_bf16.h>
#include <stdint.h>

// ---------------------------------------------------------------------------
// DIN attention pooling, fused, WMMA f16 (f32 accumulate) for the MLP.
// B=2048, T=200, E=128, H1=80, H2=40.  One workgroup per batch element.
// K-loop ordered as (column-chunk x segment): the keys fragment is loaded
// once per chunk and the four feature variants (q, k, q-k, q*k) are derived
// with packed f16 ops at compile time -- no runtime segment dispatch.
// ---------------------------------------------------------------------------

#define B_  2048
#define T_  200
#define TP  208            // T padded to 13 * 16
#define E_  128
#define K1  512            // 4*E
#define H1_ 80
#define H2_ 40
#define H1P 96             // H1 padded to 3*32 (K of GEMM2)
#define H2P 48             // H2 padded to 3*16 (N of GEMM2)

typedef _Float16 h16v __attribute__((ext_vector_type(16)));
typedef float    f8v  __attribute__((ext_vector_type(8)));

// ---------------------------------------------------------------------------
// Prep: W1 (512x80 f32, row major) -> W1t f16 [80][512]
//       W2 (80x40  f32, row major) -> W2t f16 [48][96] zero padded
// ---------------------------------------------------------------------------
__global__ void din_prep_kernel(const float* __restrict__ W1,
                                const float* __restrict__ W2,
                                _Float16* __restrict__ w1t,
                                _Float16* __restrict__ w2t) {
    int i = blockIdx.x * 256 + threadIdx.x;
    if (i < H1_ * K1) {                       // 40960
        int n = i / K1, k = i % K1;
        w1t[n * K1 + k] = (_Float16)W1[k * H1_ + n];
    }
    int j = i - H1_ * K1;
    if (j >= 0 && j < H2P * H1P) {            // 4608
        int n = j / H1P, k = j % H1P;
        _Float16 v = (_Float16)0.f;
        if (n < H2_ && k < H1_) v = (_Float16)W2[k * H2_ + n];
        w2t[j] = v;
    }
}

// Load a 16x32 f16 A-fragment row slice (A layout: lane<16 holds K offsets
// {0..15}, lane>=16 holds {16..31}) from a contiguous row.
__device__ __forceinline__ h16v load_afrag(const _Float16* __restrict__ row,
                                           int kbase, int hg) {
    h16v a;
    const _Float16* p0 = row + kbase + hg * 16;
    const _Float16* p1 = row + kbase + hg * 16 + 8;
#pragma unroll
    for (int i = 0; i < 8; ++i) { a[i] = p0[i]; a[i + 8] = p1[i]; }
    return a;
}

// Load a 32x16 f16 B-fragment (lane<16 holds K 0..15, lane>=16 holds K 16..31,
// N = lane%16) from a row-per-N, K-contiguous LDS layout.
__device__ __forceinline__ h16v load_bfrag(const _Float16* __restrict__ p) {
    h16v bf;
#pragma unroll
    for (int i = 0; i < 16; ++i) bf[i] = p[i];
    return bf;
}

__global__ __launch_bounds__(256, 1)
void din_attention_pool_kernel(const float* __restrict__ query,       // (B,1,E)
                               const float* __restrict__ keys,        // (B,T,E)
                               const unsigned char* __restrict__ mask,// (B,1,T) bool
                               const _Float16* __restrict__ w1t,      // [80][512]
                               const _Float16* __restrict__ w2t,      // [48][96]
                               const float* __restrict__ b1,
                               const float* __restrict__ b2,
                               const float* __restrict__ W3,
                               const float* __restrict__ b3,
                               float* __restrict__ out) {             // (B,1,E)
    __shared__ __align__(16) _Float16 s_w1t[H1_ * K1];      //  81920 B
    __shared__ __align__(16) _Float16 s_w2t[H2P * H1P];     //   9216 B
    __shared__ __align__(16) _Float16 s_keys[TP * E_];      //  53248 B
    __shared__ __align__(16) _Float16 s_h1[TP * H1P];       //  39936 B
    __shared__ __align__(16) _Float16 s_q[E_];              //    256 B
    __shared__ __align__(16) float    s_sc[TP];             //    832 B
    __shared__ float s_b1[H1_];
    __shared__ float s_b2[H2_];
    __shared__ float s_w3[H2_];

    const int tid  = threadIdx.x;
    const int lane = tid & 31;
    const int wave = tid >> 5;                // 8 waves
    const int b    = blockIdx.x;

    // ---------------- stage LDS (vectorized) ----------------
    {
        const uint4* src = (const uint4*)w1t;              // 5120 uint4
        uint4* dst = (uint4*)s_w1t;
#pragma unroll 4
        for (int i = tid; i < (H1_ * K1) / 8; i += 256) dst[i] = src[i];

        const uint4* src2 = (const uint4*)w2t;             // 576 uint4
        uint4* dst2 = (uint4*)s_w2t;
        for (int i = tid; i < (H2P * H1P) / 8; i += 256) dst2[i] = src2[i];

        const float4* kb4 = (const float4*)(keys + (size_t)b * (T_ * E_));
        uint2* kd = (uint2*)s_keys;
#pragma unroll 4
        for (int i = tid; i < (T_ * E_) / 4; i += 256) {   // 6400 float4
            float4 v = kb4[i];
            union { _Float16 h[4]; uint2 u; } p;
            p.h[0] = (_Float16)v.x; p.h[1] = (_Float16)v.y;
            p.h[2] = (_Float16)v.z; p.h[3] = (_Float16)v.w;
            kd[i] = p.u;
        }
        for (int i = (T_ * E_) / 4 + tid; i < (TP * E_) / 4; i += 256)
            kd[i] = make_uint2(0u, 0u);                    // pad rows 200..207

        uint2* hz = (uint2*)s_h1;
        for (int i = tid; i < (TP * H1P) / 4; i += 256) hz[i] = make_uint2(0u, 0u);

        if (tid < E_)  s_q[tid]  = (_Float16)query[(size_t)b * E_ + tid];
        if (tid < H1_) s_b1[tid] = b1[tid];
        if (tid < H2_) { s_b2[tid] = b2[tid]; s_w3[tid] = W3[tid]; }
    }
    __syncthreads();

    const int hg   = lane >> 4;     // half-wave group (0/1)
    const int lrow = lane & 15;

    // q fragments per column chunk do not depend on the row: hoist them.
    h16v aq[4];
#pragma unroll
    for (int cc = 0; cc < 4; ++cc) aq[cc] = load_afrag(s_q, cc * 32, hg);

    // ---------------- GEMM1: feat(208x512) @ W1(512x80) -> H1 ----------------
    // Each wave owns one M-tile; 5 N-tile accumulators share every A-fragment.
    // Chunk kc = seg*4 + cc; segment fragments derived from one keys load.
    for (int mt = wave; mt < 13; mt += 8) {
        const _Float16* krow = s_keys + (mt * 16 + lrow) * E_;
        f8v acc[5];
#pragma unroll
        for (int nt = 0; nt < 5; ++nt) acc[nt] = (f8v){0.f,0.f,0.f,0.f,0.f,0.f,0.f,0.f};
#pragma unroll
        for (int cc = 0; cc < 4; ++cc) {
            const h16v ak  = load_afrag(krow, cc * 32, hg);
            const h16v amk = aq[cc] - ak;        // packed f16
            const h16v axk = aq[cc] * ak;        // packed f16
#pragma unroll
            for (int s = 0; s < 4; ++s) {
                const h16v a = (s == 0) ? aq[cc] : (s == 1) ? ak
                             : (s == 2) ? amk    : axk;
                const int kc = s * 4 + cc;
#pragma unroll
                for (int nt = 0; nt < 5; ++nt) {
                    const h16v bf = load_bfrag(s_w1t + (nt * 16 + lrow) * K1
                                               + kc * 32 + hg * 16);
                    acc[nt] = __builtin_amdgcn_wmma_f32_16x16x32_f16(
                                  false, a, false, bf, (short)0, acc[nt],
                                  false, false);
                }
            }
        }
        // D layout: vgpr i -> M = i + 8*hg, lane%16 -> N
#pragma unroll
        for (int nt = 0; nt < 5; ++nt) {
            const int   n   = nt * 16 + lrow;
            const float b1v = s_b1[n];
#pragma unroll
            for (int i = 0; i < 8; ++i) {
                float v = acc[nt][i] + b1v;
                v = v > 0.f ? v : 0.f;
                s_h1[(mt * 16 + hg * 8 + i) * H1P + n] = (_Float16)v;
            }
        }
    }
    __syncthreads();

    // ---------------- GEMM2: H1(208x96) @ W2t(96x48), fold W3 ----------------
    // Each wave owns one M-tile across all 3 N-tiles -> per-row score via
    // in-wave shuffle reduction, single LDS store (no atomics).
    for (int mt = wave; mt < 13; mt += 8) {
        const _Float16* arow = s_h1 + (mt * 16 + lrow) * H1P;
        f8v acc[3];
#pragma unroll
        for (int nt = 0; nt < 3; ++nt) acc[nt] = (f8v){0.f,0.f,0.f,0.f,0.f,0.f,0.f,0.f};
#pragma unroll
        for (int kc = 0; kc < 3; ++kc) {
            const h16v a = load_afrag(arow, kc * 32, hg);
#pragma unroll
            for (int nt = 0; nt < 3; ++nt) {
                const h16v bf = load_bfrag(s_w2t + (nt * 16 + lrow) * H1P
                                           + kc * 32 + hg * 16);
                acc[nt] = __builtin_amdgcn_wmma_f32_16x16x32_f16(
                              false, a, false, bf, (short)0, acc[nt],
                              false, false);
            }
        }
        float part[8];
#pragma unroll
        for (int i = 0; i < 8; ++i) part[i] = 0.f;
#pragma unroll
        for (int nt = 0; nt < 3; ++nt) {
            const int   n   = nt * 16 + lrow;
            const float w3v = (n < H2_) ? s_w3[n] : 0.f;
            const float b2v = (n < H2_) ? s_b2[n] : 0.f;
#pragma unroll
            for (int i = 0; i < 8; ++i) {
                float v = acc[nt][i] + b2v;
                v = v > 0.f ? v : 0.f;
                part[i] = fmaf(v, w3v, part[i]);
            }
        }
        // reduce across the 16 lanes holding different n (width-16 subgroups)
#pragma unroll
        for (int m = 1; m < 16; m <<= 1) {
#pragma unroll
            for (int i = 0; i < 8; ++i) part[i] += __shfl_xor(part[i], m, 16);
        }
        if (lrow == 0) {
#pragma unroll
            for (int i = 0; i < 8; ++i) s_sc[mt * 16 + hg * 8 + i] = part[i];
        }
    }
    __syncthreads();

    // ---------------- mask + b3 ----------------
    if (tid < TP) {
        float s = 0.f;
        if (tid < T_) {
            unsigned char m = mask[(size_t)b * T_ + tid];
            s = m ? (s_sc[tid] + b3[0]) : 0.f;
        }
        s_sc[tid] = s;
    }
    __syncthreads();

    // ---------------- pooling: out[b][e] = sum_t s[t] * keys[t][e] ----------
    if (tid < E_) {
        float acc = 0.f;
#pragma unroll 8
        for (int t = 0; t < T_; ++t)
            acc = fmaf(s_sc[t], (float)s_keys[t * E_ + tid], acc);
        out[(size_t)b * E_ + tid] = acc;
    }
}

extern "C" void kernel_launch(void* const* d_in, const int* in_sizes, int n_in,
                              void* d_out, int out_size, void* d_ws, size_t ws_size,
                              hipStream_t stream) {
    const float*         query = (const float*)d_in[0];
    const float*         keys  = (const float*)d_in[1];
    const unsigned char* kmask = (const unsigned char*)d_in[2];  // jnp.bool_ -> 1 byte
    const float*         W1    = (const float*)d_in[3];
    const float*         b1    = (const float*)d_in[4];
    const float*         W2    = (const float*)d_in[5];
    const float*         b2    = (const float*)d_in[6];
    const float*         W3    = (const float*)d_in[7];
    const float*         b3    = (const float*)d_in[8];
    float*               out   = (float*)d_out;

    _Float16* w1t = (_Float16*)d_ws;                  // 40960 halves
    _Float16* w2t = w1t + H1_ * K1;                   // 4608 halves

    const int prep_elems = H1_ * K1 + H2P * H1P;      // 45568
    din_prep_kernel<<<(prep_elems + 255) / 256, 256, 0, stream>>>(W1, W2, w1t, w2t);

    din_attention_pool_kernel<<<B_, 256, 0, stream>>>(query, keys, kmask,
                                                      w1t, w2t, b1, b2, W3, b3, out);
}
